// VietaPellKANLayer_30331059044962
// MI455X (gfx1250) — matrix operand order repaired and checked by reading, not compiled, not run on hardware
//
#include <hip/hip_runtime.h>

// ---------------------------------------------------------------------------
// VietaPellKAN layer for MI455X (gfx1250, wave32, WMMA)
//   y[b,o] = sum_i sum_d V_d(tanh(x[b,i])) * C[i,o,d]
// One f16 GEMM: M=4096, N=1024, K=9*1024=9216, f32 accumulate.
//   pass 1: T = tanh(x) as f16 (bandwidth-trivial, avoids 8x recompute)
//   pass 2: repack coeffs -> f16 [o][d*1024+i], scaled 2^8 (dodge f16 denorms)
//   pass 3: fused GEMM; Vieta-Pell recurrence carried in packed-f16 registers,
//           V_d staged through a double-buffered padded LDS tile into
//           v_wmma_f32_16x16x32_f16; B fragments stream from L2-resident pack.
// ---------------------------------------------------------------------------

typedef __attribute__((ext_vector_type(16))) _Float16 v16h;
typedef __attribute__((ext_vector_type(8)))  float    v8f;
typedef __attribute__((ext_vector_type(2)))  _Float16 h2;

#define BATCH 4096
#define IDIM  1024
#define ODIM  1024
#define NDEG  9                 // degree 8 -> 9 basis terms
#define KTOT  (IDIM * NDEG)     // 9216 halfs per Bpack row
#define KC    32                // i-chunk == WMMA K
#define BM    128
#define BN    128
#define LDA   40                // padded LDS row stride (halfs): 80B, conflict-free
#define SCALE      256.0f
#define INV_SCALE  (1.0f / 256.0f)

#define BP_BYTES ((size_t)ODIM * KTOT * 2)   // 18.9 MB  (f16 coeff pack)

union FragH  { v16h h; uint4 u[2]; _Float16 e[16]; };
union Pack16 { h2 h[8]; uint4 u[2]; _Float16 e[16]; };

// --------------------------- pass 1: tanh ----------------------------------
__global__ __launch_bounds__(256) void tanh_f16_kernel(const float* __restrict__ x,
                                                       _Float16* __restrict__ T) {
    const size_t idx = ((size_t)blockIdx.x * 256 + threadIdx.x) * 8;
    float4 a = *(const float4*)(x + idx);
    float4 b = *(const float4*)(x + idx + 4);
    union { uint4 u; _Float16 h[8]; } r;
    r.h[0] = (_Float16)tanhf(a.x);  r.h[1] = (_Float16)tanhf(a.y);
    r.h[2] = (_Float16)tanhf(a.z);  r.h[3] = (_Float16)tanhf(a.w);
    r.h[4] = (_Float16)tanhf(b.x);  r.h[5] = (_Float16)tanhf(b.y);
    r.h[6] = (_Float16)tanhf(b.z);  r.h[7] = (_Float16)tanhf(b.w);
    *(uint4*)(T + idx) = r.u;
}

// --------------------------- pass 2: coeff repack --------------------------
// Bpack[o*KTOT + d*IDIM + i] = C[i*KTOT + o*9 + d] * 2^8  (f32 -> f16)
// Lane l owns i: reads stream contiguously per lane, writes are 2B/lane
// contiguous in i -> fully coalesced.
__global__ __launch_bounds__(256) void repack_coeffs(const float* __restrict__ coeffs,
                                                     _Float16* __restrict__ bp) {
    const int i  = blockIdx.x * 256 + threadIdx.x;   // 0..1023
    const int o0 = blockIdx.y * 32;
    const float* src = coeffs + (size_t)i * KTOT + (size_t)o0 * NDEG;
    for (int oo = 0; oo < 32; ++oo) {
        const int o = o0 + oo;
        #pragma unroll
        for (int d = 0; d < NDEG; ++d) {
            bp[(size_t)o * KTOT + d * IDIM + i] = (_Float16)(src[oo * NDEG + d] * SCALE);
        }
    }
}

// --------------------------- pass 3: fused WMMA GEMM -----------------------
__global__ __launch_bounds__(256) void vieta_pell_wmma(const _Float16* __restrict__ T,
                                                       const _Float16* __restrict__ bp,
                                                       float* __restrict__ y) {
    // double-buffered A tile: 128 x 32 f16, padded rows (80B stride)
    __shared__ __align__(16) _Float16 sA[2][BM * LDA];

    const int tid  = threadIdx.x;
    const int lane = tid & 31;
    const int wave = tid >> 5;        // 0..7
    const int wm   = wave >> 1;       // 0..3  -> 32-row slab
    const int wn   = wave & 1;        // 0..1  -> 64-col slab
    const int l16  = lane & 15;
    const int lhi  = lane >> 4;       // 0/1: which K-half this lane holds

    const int m0 = blockIdx.y * BM;
    const int n0 = blockIdx.x * BN;

    // recurrence ownership: thread -> row (tid>>1), 16-col half ((tid&1)*16)
    const int rrow = tid >> 1;
    const int rcol = (tid & 1) * 16;

    v8f acc[2][4] = {};

    const h2 two = { (_Float16)2.0f, (_Float16)2.0f };

    int pb = 0;
    #pragma unroll 1
    for (int ic = 0; ic < IDIM / KC; ++ic) {
        const int i0 = ic * KC;

        // ---- load tanh tile (2 x b128 per thread), keep recurrence in f16 ----
        Pack16 t, vc, vn;
        const _Float16* tp = T + (size_t)(m0 + rrow) * IDIM + i0 + rcol;
        t.u[0] = *(const uint4*)(tp);
        t.u[1] = *(const uint4*)(tp + 8);
        #pragma unroll
        for (int e = 0; e < 8; ++e) { vc.h[e] = two; vn.h[e] = t.h[e]; }  // V0=2, V1=t

        for (int d = 0; d < NDEG; ++d) {
            // ---- stage V_d into LDS (2 x ds_store_b128, already f16) ----
            _Float16* dst = &sA[pb][rrow * LDA + rcol];
            *(uint4*)(dst)     = vc.u[0];
            *(uint4*)(dst + 8) = vc.u[1];

            __syncthreads();   // writes(pb) -> reads(pb); double buffer => 1 barrier

            // ---- A fragments from LDS (contiguous-k 16x32 A layout) ----
            // lane<16 holds K 0-15; lane>=16 holds K 16-31 (contiguous, matches B)
            FragH af[2];
            #pragma unroll
            for (int mi = 0; mi < 2; ++mi) {
                const _Float16* ap = &sA[pb][(wm * 32 + mi * 16 + l16) * LDA + lhi * 16];
                af[mi].u[0] = *(const uint4*)(ap);
                af[mi].u[1] = *(const uint4*)(ap + 8);
            }

            // ---- B fragments direct from L2-resident Bpack ----
            // lane holds column n = base+l16, K = lhi*16 .. +15 (contiguous)
            FragH bf[4];
            #pragma unroll
            for (int ni = 0; ni < 4; ++ni) {
                const _Float16* bpp = bp + (size_t)(n0 + wn * 64 + ni * 16 + l16) * KTOT
                                         + d * IDIM + i0 + lhi * 16;
                bf[ni].u[0] = *(const uint4*)(bpp);
                bf[ni].u[1] = *(const uint4*)(bpp + 8);
                __builtin_prefetch((const void*)(bpp + IDIM), 0, 1);  // next degree plane
            }

            // ---- 8 x v_wmma_f32_16x16x32_f16 ----
            #pragma unroll
            for (int mi = 0; mi < 2; ++mi)
                #pragma unroll
                for (int ni = 0; ni < 4; ++ni)
                    acc[mi][ni] = __builtin_amdgcn_wmma_f32_16x16x32_f16(
                        false, af[mi].h, false, bf[ni].h,
                        (short)0, acc[mi][ni], false, false);

            // ---- Vieta-Pell advance in packed f16: V_{d+2} = t*V_{d+1} + V_d ----
            #pragma unroll
            for (int e = 0; e < 8; ++e) {
                h2 nx = t.h[e] * vn.h[e] + vc.h[e];   // -> v_pk_fma_f16
                vc.h[e] = vn.h[e];
                vn.h[e] = nx;
            }
            pb ^= 1;
        }
    }

    // ---- epilogue: C layout row = base + v + 8*lhi, col = base + l16 ----
    #pragma unroll
    for (int mi = 0; mi < 2; ++mi) {
        #pragma unroll
        for (int ni = 0; ni < 4; ++ni) {
            const int row = m0 + wm * 32 + mi * 16 + lhi * 8;
            const int col = n0 + wn * 64 + ni * 16 + l16;
            float* op = y + (size_t)row * ODIM + col;
            #pragma unroll
            for (int v = 0; v < 8; ++v)
                op[(size_t)v * ODIM] = acc[mi][ni][v] * INV_SCALE;
        }
    }
}

// ---------------------------------------------------------------------------
extern "C" void kernel_launch(void* const* d_in, const int* in_sizes, int n_in,
                              void* d_out, int out_size, void* d_ws, size_t ws_size,
                              hipStream_t stream) {
    (void)in_sizes; (void)n_in; (void)out_size; (void)ws_size;
    const float* x      = (const float*)d_in[0];   // [4096,1024] f32
    const float* coeffs = (const float*)d_in[1];   // [1024,1024,9] f32
    float*       yout   = (float*)d_out;           // [4096,1024] f32

    _Float16* bpack = (_Float16*)d_ws;                           // 18.9 MB
    _Float16* tbuf  = (_Float16*)((char*)d_ws + BP_BYTES);       // + 8.0 MB

    tanh_f16_kernel<<<(BATCH * IDIM) / (256 * 8), 256, 0, stream>>>(x, tbuf);
    repack_coeffs<<<dim3(IDIM / 256, ODIM / 32), 256, 0, stream>>>(coeffs, bpack);
    vieta_pell_wmma<<<dim3(ODIM / BN, BATCH / BM), 256, 0, stream>>>(tbuf, bpack, yout);
}
